// MambaBlock_36636071035407
// MI455X (gfx1250) — matrix operand order repaired and checked by reading, not compiled, not run on hardware
//
#include <hip/hip_runtime.h>
#include <hip/hip_bf16.h>
#include <stdint.h>

// ---------------------------------------------------------------------------
// Mamba block forward for MI455X (gfx1250, wave32, WMMA).
// GEMMs: v_wmma_f32_16x16x32_bf16, bf16 operands staged in LDS via
// global_load_async_to_lds_b128 (ASYNCcnt path) with fp32 accumulate.
// All fp32->bf16 conversion is done once, outside the GEMM hot loops.
// ---------------------------------------------------------------------------

typedef __attribute__((ext_vector_type(16))) __bf16 v16bf;
typedef __attribute__((ext_vector_type(8)))  __bf16 v8bf;
typedef __attribute__((ext_vector_type(8)))  float  v8f;

#if defined(__has_builtin)
#if __has_builtin(__builtin_amdgcn_global_load_async_to_lds_b128) && \
    __has_builtin(__builtin_amdgcn_s_wait_asynccnt)
#define USE_ASYNC_LDS 1
#endif
#endif

__device__ __forceinline__ uint32_t bfbits(float f) {
    union { float f; uint32_t u; } v; v.f = f;
    return (v.u + 0x7FFFu + ((v.u >> 16) & 1u)) >> 16;     // round-to-nearest-even
}

__device__ __forceinline__ __bf16 f2bf(float f) {
    uint16_t h = (uint16_t)bfbits(f);
    __bf16 b;
    __builtin_memcpy(&b, &h, 2);
    return b;
}

__device__ __forceinline__ float silu_f(float x) {
    return x / (1.0f + __expf(-x));
}

// 16-byte global -> LDS copy (async path when toolchain exposes it)
__device__ __forceinline__ void copy16_g2l(const __bf16* g, __bf16* l) {
#if defined(USE_ASYNC_LDS)
    typedef int v4i __attribute__((vector_size(16)));
    __builtin_amdgcn_global_load_async_to_lds_b128(
        (__attribute__((address_space(1))) v4i*)g,
        (__attribute__((address_space(3))) v4i*)l, 0, 0);
#else
    *(uint4*)l = *(const uint4*)g;
#endif
}

__device__ __forceinline__ void wait_stage() {
#if defined(USE_ASYNC_LDS)
    __builtin_amdgcn_s_wait_asynccnt(0);
#endif
    __syncthreads();
}

// ---------------------------------------------------------------------------
// fp32 -> bf16 pack (4 elements / thread)
// ---------------------------------------------------------------------------
__global__ __launch_bounds__(256) void pack_bf16_kernel(
    const float* __restrict__ in, __bf16* __restrict__ out, int n4)
{
    const int i = blockIdx.x * 256 + threadIdx.x;
    if (i >= n4) return;
    const float4 v = ((const float4*)in)[i];
    uint2 r;
    r.x = bfbits(v.x) | (bfbits(v.y) << 16);
    r.y = bfbits(v.z) | (bfbits(v.w) << 16);
    ((uint2*)out)[i] = r;
}

// ---------------------------------------------------------------------------
// NT GEMM: C[m,n] = sum_k A[m,k]*B[n,k], A/B bf16 row-major (K contiguous).
// EPI:  0 = none, 1 = softplus(bias[n] + acc)
// OUTF: 0 = fp32 store, 1 = bf16 store
// 256 threads = 8 waves (WM x WN), each wave TM x TN 16x16 WMMA tiles.
// Requires M%BM==0, N%BN==0, K%32==0 (true for all calls).
// ---------------------------------------------------------------------------
template <int BM, int BN, int WM, int WN, int EPI, int OUTF>
__global__ __launch_bounds__(256) void gemm_nt_bf16(
    const __bf16* __restrict__ A, const __bf16* __restrict__ B,
    void* __restrict__ Cout, const float* __restrict__ bias,
    int N, int K)
{
    static_assert(WM * WN == 8, "8 waves per block");
    constexpr int KT  = 32;
    constexpr int LST = KT + 8;          // LDS row stride (bf16), keeps 16B align
    constexpr int TM  = BM / (16 * WM);
    constexpr int TN  = BN / (16 * WN);
    constexpr int ACH = BM * 4;          // 16B chunks in A tile
    constexpr int BCH = BN * 4;
    constexpr int AREP = (ACH + 255) / 256;
    constexpr int BREP = (BCH + 255) / 256;
    constexpr bool AFULL = (ACH % 256) == 0;   // all threads active every rep
    constexpr bool BFULL = (BCH % 256) == 0;

    __shared__ __bf16 sA[BM * LST];
    __shared__ __bf16 sB[BN * LST];

    const int tid  = threadIdx.x;
    const int wave = tid >> 5;
    const int lane = tid & 31;
    const int lg   = lane >> 4;
    const int ln   = lane & 15;
    const int wm   = wave % WM;
    const int wn   = wave / WM;
    const int blockM = blockIdx.y * BM;
    const int blockN = blockIdx.x * BN;

    // ---- precompute staging pointers (advance by KT each K-step) ----
    const __bf16* agp[AREP]; __bf16* alp[AREP]; bool aact[AREP];
#pragma unroll
    for (int r2 = 0; r2 < AREP; ++r2) {
        const int idx = tid + r2 * 256;
        aact[r2] = AFULL || (idx < ACH);
        const int rr = aact[r2] ? (idx >> 2) : 0;
        const int cc = idx & 3;
        agp[r2] = A + (size_t)(blockM + rr) * K + cc * 8;
        alp[r2] = &sA[rr * LST + cc * 8];
    }
    const __bf16* bgp[BREP]; __bf16* blp[BREP]; bool bact[BREP];
#pragma unroll
    for (int r2 = 0; r2 < BREP; ++r2) {
        const int idx = tid + r2 * 256;
        bact[r2] = BFULL || (idx < BCH);
        const int rr = bact[r2] ? (idx >> 2) : 0;
        const int cc = idx & 3;
        bgp[r2] = B + (size_t)(blockN + rr) * K + cc * 8;
        blp[r2] = &sB[rr * LST + cc * 8];
    }

    v8f acc[TM][TN];
#pragma unroll
    for (int i = 0; i < TM; ++i)
#pragma unroll
        for (int j = 0; j < TN; ++j)
#pragma unroll
            for (int e = 0; e < 8; ++e) acc[i][j][e] = 0.0f;

    for (int k0 = 0; k0 < K; k0 += KT) {
        // ---- stage tiles (compile-time unconditional for full configs) ----
#pragma unroll
        for (int r2 = 0; r2 < AREP; ++r2)
            if (AFULL || aact[r2]) { copy16_g2l(agp[r2], alp[r2]); agp[r2] += KT; }
#pragma unroll
        for (int r2 = 0; r2 < BREP; ++r2)
            if (BFULL || bact[r2]) { copy16_g2l(bgp[r2], blp[r2]); bgp[r2] += KT; }

        if (k0 + KT < K) {   // prefetch next K-tile into cache hierarchy
#pragma unroll
            for (int r2 = 0; r2 < AREP; ++r2)
                if (AFULL || aact[r2]) __builtin_prefetch(agp[r2], 0, 0);
#pragma unroll
            for (int r2 = 0; r2 < BREP; ++r2)
                if (BFULL || bact[r2]) __builtin_prefetch(bgp[r2], 0, 0);
        }
        wait_stage();

        // ---- fragments per ISA wave32 VGPR layouts ----
        // A 16x32: row m = ln; j<8 -> k=8*lg+j ; j>=8 -> k=16+8*lg+(j-8)
        v16bf afrag[TM];
#pragma unroll
        for (int i = 0; i < TM; ++i) {
            const int m = wm * 16 * TM + i * 16 + ln;
            const v8bf lo = *(const v8bf*)&sA[m * LST + 8 * lg];
            const v8bf hi = *(const v8bf*)&sA[m * LST + 16 + 8 * lg];
#pragma unroll
            for (int e = 0; e < 8; ++e) { afrag[i][e] = lo[e]; afrag[i][e + 8] = hi[e]; }
        }
        // B 32x16: col n = ln; j<8 -> k=8*lg+j ; j>=8 -> k=16+8*lg+(j-8)
        v16bf bfrag[TN];
#pragma unroll
        for (int j = 0; j < TN; ++j) {
            const int n = wn * 16 * TN + j * 16 + ln;
            const v8bf lo = *(const v8bf*)&sB[n * LST + 8 * lg];
            const v8bf hi = *(const v8bf*)&sB[n * LST + 16 + 8 * lg];
#pragma unroll
            for (int e = 0; e < 8; ++e) { bfrag[j][e] = lo[e]; bfrag[j][e + 8] = hi[e]; }
        }

#pragma unroll
        for (int i = 0; i < TM; ++i)
#pragma unroll
            for (int j = 0; j < TN; ++j)
                acc[i][j] = __builtin_amdgcn_wmma_f32_16x16x32_bf16(
                    false, afrag[i], false, bfrag[j],
                    (short)0, acc[i][j], false, false);

        __syncthreads();
    }

    // ---- epilogue: C/D layout: elem v -> m = v + 8*lg, col n = ln ----
#pragma unroll
    for (int i = 0; i < TM; ++i) {
#pragma unroll
        for (int j = 0; j < TN; ++j) {
            const int n     = blockN + wn * 16 * TN + j * 16 + ln;
            const int mBase = blockM + wm * 16 * TM + i * 16 + 8 * lg;
            float bn = 0.0f;
            if (EPI == 1) bn = bias[n];
#pragma unroll
            for (int v = 0; v < 8; ++v) {
                float x = acc[i][j][v];
                if (EPI == 1) {
                    x += bn;
                    x = (x > 20.0f) ? x : log1pf(__expf(x));   // softplus
                }
                if (OUTF == 0) ((float*)Cout)[(size_t)(mBase + v) * N + n] = x;
                else           ((__bf16*)Cout)[(size_t)(mBase + v) * N + n] = f2bf(x);
            }
        }
    }
}

// ---------------------------------------------------------------------------
// Depthwise causal conv (K=4) + bias + SiLU; emits fp32 (for scan) and bf16
// (for downstream WMMA GEMMs).  ab: [2048,4096], a = cols 0..2047.
// ---------------------------------------------------------------------------
__global__ __launch_bounds__(256) void conv_silu_kernel(
    const float* __restrict__ ab, const float* __restrict__ cw,
    const float* __restrict__ cb, float* __restrict__ aconv,
    __bf16* __restrict__ aconv_bf)
{
    const int idx = blockIdx.x * 256 + threadIdx.x;     // over 2048*2048
    const int c = idx & 2047;
    const int m = idx >> 11;
    const int t = m & 1023;                             // position within batch

    float acc = cb[c];
#pragma unroll
    for (int k = 0; k < 4; ++k) {
        const int tt = t - 3 + k;
        if (tt >= 0)
            acc += ab[(size_t)(m - 3 + k) * 4096 + c] * cw[c * 4 + k];
    }
    const float s = silu_f(acc);
    aconv[(size_t)m * 2048 + c]    = s;
    aconv_bf[(size_t)m * 2048 + c] = f2bf(s);
}

// ---------------------------------------------------------------------------
// Sequential SSM scan: one lane per (batch, channel); 16 states in registers.
// y[m,d] = (sum_s h_s*C_s + D[d]*a) * silu(g)   (y stored bf16 for final GEMM)
// ---------------------------------------------------------------------------
__global__ __launch_bounds__(256) void ssm_scan_kernel(
    const float* __restrict__ sdelta, const float* __restrict__ aconv,
    const float* __restrict__ ab,     const float* __restrict__ Bb,
    const float* __restrict__ Cb,     const float* __restrict__ A_param,
    const float* __restrict__ D_param, __bf16* __restrict__ ybf)
{
    const int gid = blockIdx.x * 256 + threadIdx.x;     // 0..4095
    const int d  = gid & 2047;
    const int bb = gid >> 11;

    float eA[16];
#pragma unroll
    for (int s = 0; s < 16; ++s) eA[s] = __expf(-A_param[d * 16 + s]);
    const float Dp = D_param[d];

    float h[16];
#pragma unroll
    for (int s = 0; s < 16; ++s) h[s] = 0.0f;

    for (int t = 0; t < 1024; ++t) {
        const int m = bb * 1024 + t;
        const float dl = sdelta[(size_t)m * 2048 + d];
        const float av = aconv[(size_t)m * 2048 + d];
        const float gv = ab[(size_t)m * 4096 + 2048 + d];
        const float dav = dl * av;
        const float* __restrict__ Bp = Bb + (size_t)m * 16;
        const float* __restrict__ Cp = Cb + (size_t)m * 16;
        float dot = 0.0f;
#pragma unroll
        for (int s = 0; s < 16; ++s) {
            h[s] = eA[s] * dl * h[s] + Bp[s] * dav;     // h = A_bar*h + X_bar
            dot += h[s] * Cp[s];
        }
        ybf[(size_t)m * 2048 + d] = f2bf((dot + Dp * av) * silu_f(gv));
    }
}

// ---------------------------------------------------------------------------
// Launch sequence
// ---------------------------------------------------------------------------
extern "C" void kernel_launch(void* const* d_in, const int* in_sizes, int n_in,
                              void* d_out, int out_size, void* d_ws, size_t ws_size,
                              hipStream_t stream)
{
    (void)in_sizes; (void)n_in; (void)out_size; (void)ws_size;

    const float* seq    = (const float*)d_in[0];   // [2,1024,1024]
    const float* W_in   = (const float*)d_in[1];   // [4096,1024]
    const float* W_out  = (const float*)d_in[2];   // [1024,2048]
    const float* W_sB   = (const float*)d_in[3];   // [16,2048]
    const float* W_sC   = (const float*)d_in[4];   // [16,2048]
    const float* W_sD1  = (const float*)d_in[5];   // [128,2048]
    const float* W_sD2  = (const float*)d_in[6];   // [2048,128]
    const float* conv_w = (const float*)d_in[7];   // [2048,1,4]
    const float* conv_b = (const float*)d_in[8];   // [2048]
    const float* A_par  = (const float*)d_in[9];   // [2048,16]
    const float* D_par  = (const float*)d_in[10];  // [2048]
    float* out = (float*)d_out;                    // [2,1024,1024]

    char* ws = (char*)d_ws;
    float*  ab      = (float*)(ws + 0);                    // 2048*4096 f32 = 32 MB
    float*  aconv   = (float*)(ws + (size_t) 33554432);    // 2048*2048 f32 = 16 MB
    float*  sdelta  = (float*)(ws + (size_t) 50331648);    // 2048*2048 f32 = 16 MB
    __bf16* aconvbf = (__bf16*)(ws + (size_t) 67108864);   // 2048*2048 bf16 = 8 MB
    __bf16* ybf     = (__bf16*)(ws + (size_t) 75497472);   // 2048*2048 bf16 = 8 MB
    __bf16* seqbf   = (__bf16*)(ws + (size_t) 83886080);   // 2048*1024 bf16 = 4 MB
    __bf16* Winbf   = (__bf16*)(ws + (size_t) 88080384);   // 4096*1024 bf16 = 8 MB
    __bf16* Woutbf  = (__bf16*)(ws + (size_t) 96468992);   // 1024*2048 bf16 = 4 MB
    __bf16* WsD1bf  = (__bf16*)(ws + (size_t)100663296);   // 128*2048 bf16
    __bf16* WsD2bf  = (__bf16*)(ws + (size_t)101187584);   // 2048*128 bf16
    __bf16* WsBbf   = (__bf16*)(ws + (size_t)101711872);   // 16*2048 bf16
    __bf16* WsCbf   = (__bf16*)(ws + (size_t)101777408);   // 16*2048 bf16
    __bf16* d1bf    = (__bf16*)(ws + (size_t)101842944);   // 2048*128 bf16
    float*  Bbuf    = (float*)(ws + (size_t)102367232);    // 2048*16 f32
    float*  Cbuf    = (float*)(ws + (size_t)102498304);    // 2048*16 f32

    // 0) one-time fp32 -> bf16 packing (weights + seq)
    pack_bf16_kernel<<< 524288 / 256, 256, 0, stream>>>(seq,   seqbf,  524288);
    pack_bf16_kernel<<<1048576 / 256, 256, 0, stream>>>(W_in,  Winbf, 1048576);
    pack_bf16_kernel<<< 524288 / 256, 256, 0, stream>>>(W_out, Woutbf, 524288);
    pack_bf16_kernel<<<   8192 / 256, 256, 0, stream>>>(W_sB,  WsBbf,   8192);
    pack_bf16_kernel<<<   8192 / 256, 256, 0, stream>>>(W_sC,  WsCbf,   8192);
    pack_bf16_kernel<<<  65536 / 256, 256, 0, stream>>>(W_sD1, WsD1bf, 65536);
    pack_bf16_kernel<<<  65536 / 256, 256, 0, stream>>>(W_sD2, WsD2bf, 65536);

    // 1) ab = seq @ W_in^T        M=2048 N=4096 K=1024
    gemm_nt_bf16<128, 128, 4, 2, 0, 0><<<dim3(4096 / 128, 2048 / 128), 256, 0, stream>>>(
        seqbf, Winbf, ab, nullptr, 4096, 1024);

    // 2) a = silu(causal_conv(a) + b)
    conv_silu_kernel<<<(2048 * 2048) / 256, 256, 0, stream>>>(
        ab, conv_w, conv_b, aconv, aconvbf);

    // 3) B = a @ W_sB^T           M=2048 N=16 K=2048
    gemm_nt_bf16<128, 16, 8, 1, 0, 0><<<dim3(1, 2048 / 128), 256, 0, stream>>>(
        aconvbf, WsBbf, Bbuf, nullptr, 16, 2048);

    // 4) C = a @ W_sC^T
    gemm_nt_bf16<128, 16, 8, 1, 0, 0><<<dim3(1, 2048 / 128), 256, 0, stream>>>(
        aconvbf, WsCbf, Cbuf, nullptr, 16, 2048);

    // 5) d1 = a @ W_sD1^T         M=2048 N=128 K=2048  (bf16 out)
    gemm_nt_bf16<128, 128, 4, 2, 0, 1><<<dim3(1, 2048 / 128), 256, 0, stream>>>(
        aconvbf, WsD1bf, d1bf, nullptr, 128, 2048);

    // 6) sdelta = softplus(D + d1 @ W_sD2^T)   M=2048 N=2048 K=128
    gemm_nt_bf16<128, 128, 4, 2, 1, 0><<<dim3(2048 / 128, 2048 / 128), 256, 0, stream>>>(
        d1bf, WsD2bf, sdelta, D_par, 2048, 128);

    // 7) recurrent scan -> y = (ssm + D*a) * silu(g)  (bf16 out)
    ssm_scan_kernel<<<4096 / 256, 256, 0, stream>>>(
        sdelta, aconv, ab, Bbuf, Cbuf, A_par, D_par, ybf);

    // 8) out = y @ W_out^T        M=2048 N=1024 K=2048
    gemm_nt_bf16<128, 128, 4, 2, 0, 0><<<dim3(1024 / 128, 2048 / 128), 256, 0, stream>>>(
        ybf, Woutbf, out, nullptr, 1024, 2048);
}
